// DeepseekV3MoECalibrate_45088566673494
// MI455X (gfx1250) — compile-verified, matrix operand order checked by reading, not run on hardware
//
#include <hip/hip_runtime.h>
#include <hip/hip_bf16.h>

// ---------------- problem constants ----------------
constexpr int HID   = 2048;
constexpr int N_EXP = 32;
constexpr int F_EXP = 768;
constexpr int TOP_K = 6;
constexpr int F_SH  = 1536;
constexpr int T_TOK = 1024;   // B*S

// ---------------- types ----------------
typedef __bf16 BF16;
typedef __attribute__((ext_vector_type(16))) BF16  v16bf;
typedef __attribute__((ext_vector_type(8)))  BF16  v8bf;
typedef __attribute__((ext_vector_type(8)))  float v8f;
typedef __attribute__((ext_vector_type(4)))  unsigned int u32x4;
typedef __attribute__((ext_vector_type(8)))  int i32x8;
typedef __attribute__((ext_vector_type(4)))  int i32x4;

// native f32 -> bf16 convert (backend emits v_cvt_*bf16_f32)
__device__ __forceinline__ BF16 to_bf16(float f) { return (BF16)f; }

__device__ __forceinline__ v8bf cvt8(float4 a, float4 b) {
  v8bf r;
  r[0] = (BF16)a.x; r[1] = (BF16)a.y; r[2] = (BF16)a.z; r[3] = (BF16)a.w;
  r[4] = (BF16)b.x; r[5] = (BF16)b.y; r[6] = (BF16)b.z; r[7] = (BF16)b.w;
  return r;
}

__device__ __forceinline__ v8f wmma_bf16(v16bf a, v16bf b, v8f c) {
  return __builtin_amdgcn_wmma_f32_16x16x32_bf16(false, a, false, b, (short)0, c,
                                                 false, false);
}

// ------------- Tensor Data Mover: bf16 tile [tileRows x 32] -> LDS -------------
// D# built per CDNA5 ISA (group0: count/lds/global/type,
// group1: data_size=2B, tensor dims/strides, tile dims). 2D tensor -> groups 2/3 zero.
#if defined(__clang_major__) && (__clang_major__ >= 23)
#define TDM_LOAD(g0, g1) \
  __builtin_amdgcn_tensor_load_to_lds((g0), (g1), (i32x4)0, (i32x4)0, (i32x8)0, 0)
#else
#define TDM_LOAD(g0, g1) \
  __builtin_amdgcn_tensor_load_to_lds((g0), (g1), (i32x4)0, (i32x4)0, 0)
#endif

__device__ __forceinline__ void tdm_load_bf16_tile(BF16* ldsDst, const BF16* gsrc,
                                                   unsigned tensorW /*elems*/,
                                                   unsigned tensorH /*rows*/,
                                                   unsigned tileRows) {
  unsigned long long ga = (unsigned long long)(uintptr_t)gsrc;
  unsigned lds = (unsigned)(uintptr_t)ldsDst;   // addr[31:0] = LDS offset
  u32x4 g0;
  g0[0] = 1u;                                           // count=1, no gather
  g0[1] = lds;                                          // lds_addr
  g0[2] = (unsigned)(ga & 0xFFFFFFFFull);               // global_addr[31:0]
  g0[3] = (unsigned)((ga >> 32) & 0x01FFFFFFull) | 0x80000000u;  // addr[56:32] | type=2
  i32x8 g1;
  g1[0] = (int)(1u << 16);                              // wg_mask=0, data_size=1 (2B)
  g1[1] = (int)((tensorW & 0xFFFFu) << 16);             // tensor_dim0[15:0] @bits63:48
  g1[2] = (int)((tensorW >> 16) | ((tensorH & 0xFFFFu) << 16)); // dim0 hi | dim1 lo
  g1[3] = (int)((tensorH >> 16) | (32u << 16));         // dim1 hi | tile_dim0=32
  g1[4] = (int)(tileRows & 0xFFFFu);                    // tile_dim1 | tile_dim2=0
  g1[5] = (int)tensorW;                                 // tensor_dim0_stride[31:0]
  g1[6] = 0;                                            // stride hi | dim1_stride lo
  g1[7] = 0;
  TDM_LOAD(g0, g1);
}

// ------------- LDS tile stagers (blockDim.x == 256, tile = 128 x 32) -------------

// f32 global [128 rows x 32 cols] (row stride ld) -> bf16 LDS [128][32]
__device__ __forceinline__ void g2l_f32(BF16* dst, const float* src, int ld) {
  int e   = threadIdx.x * 16;            // 256 threads * 16 elem = 4096
  int row = e >> 5;
  int col = e & 31;                      // 0 or 16
  const float4* s = (const float4*)(src + (size_t)row * ld + col);
  v8bf* d = (v8bf*)(dst + row * 32 + col);
  d[0] = cvt8(s[0], s[1]);
  d[1] = cvt8(s[2], s[3]);
}

// bf16 global [128 x 32] (row stride ld) -> bf16 LDS [128][32]
__device__ __forceinline__ void g2l_bf16(BF16* dst, const BF16* src, int ld) {
  int e   = threadIdx.x * 16;
  int row = e >> 5;
  int col = e & 31;
  const v8bf* s = (const v8bf*)(src + (size_t)row * ld + col);
  v8bf* d = (v8bf*)(dst + row * 32 + col);
  d[0] = s[0];
  d[1] = s[1];
}

// f32 global [32 k-rows x 128 n-cols] (row stride ld) -> bf16 LDS transposed [128 n][32 k]
__device__ __forceinline__ void g2l_f32_T(BF16* dst, const float* src, int ld) {
  int k  = threadIdx.x >> 3;             // 0..31
  int nc = (threadIdx.x & 7) * 16;       // 0..112
  const float4* s = (const float4*)(src + (size_t)k * ld + nc);
#pragma unroll
  for (int q = 0; q < 4; ++q) {
    float4 v = s[q];
    dst[(nc + 4 * q + 0) * 32 + k] = (BF16)v.x;
    dst[(nc + 4 * q + 1) * 32 + k] = (BF16)v.y;
    dst[(nc + 4 * q + 2) * 32 + k] = (BF16)v.z;
    dst[(nc + 4 * q + 3) * 32 + k] = (BF16)v.w;
  }
}

// ------------- fragment load from LDS tile [128][32] bf16 -------------
// CDNA5 16-bit A/B layout: lane<16 -> row=lane,    K {0..7, 16..23}
//                          lane>=16 -> row=lane-16, K {8..15, 24..31}
__device__ __forceinline__ v16bf frag_ld(const BF16* tile, int rcBase) {
  int lane = threadIdx.x & 31;
  int r    = rcBase + (lane & 15);
  int kb   = (lane >> 4) << 3;
  v8bf lo = *(const v8bf*)(tile + r * 32 + kb);
  v8bf hi = *(const v8bf*)(tile + r * 32 + kb + 16);
  return __builtin_shufflevector(lo, hi, 0, 1, 2, 3, 4, 5, 6, 7,
                                 8, 9, 10, 11, 12, 13, 14, 15);
}

// C layout: acc[i] -> m = i + (lane>=16 ? 8 : 0), n = lane & 15
__device__ __forceinline__ int c_row(int i) { return i + (((threadIdx.x & 31) >> 4) << 3); }
__device__ __forceinline__ int c_col()      { return threadIdx.x & 15; }

__device__ __forceinline__ float silu(float g) { return g / (1.f + __expf(-g)); }

// =====================================================================
// Kernel 0: gating.  1 wave per token, lane == expert.
// =====================================================================
__global__ __launch_bounds__(128) void gate_kernel(const float* __restrict__ x,
                                                   const float* __restrict__ gw,
                                                   float* __restrict__ combine) {
  int lane = threadIdx.x & 31;
  int t    = blockIdx.x * 4 + (threadIdx.x >> 5);
  const float* xr = x + (size_t)t * HID;

  float acc = 0.f;
  for (int d = 0; d < HID; d += 4) {
    float4 xv = *(const float4*)(xr + d);
    acc += xv.x * gw[(d + 0) * N_EXP + lane];
    acc += xv.y * gw[(d + 1) * N_EXP + lane];
    acc += xv.z * gw[(d + 2) * N_EXP + lane];
    acc += xv.w * gw[(d + 3) * N_EXP + lane];
  }
  float s = 1.f / (1.f + __expf(-acc));

  unsigned avail = 0xFFFFFFFFu;
  unsigned sel   = 0;
  float    sum   = 0.f;
#pragma unroll
  for (int it = 0; it < TOP_K; ++it) {
    float v = ((avail >> lane) & 1u) ? s : -1e30f;
#pragma unroll
    for (int off = 16; off > 0; off >>= 1) v = fmaxf(v, __shfl_xor(v, off, 32));
    unsigned long long b = __ballot(((avail >> lane) & 1u) && (s == v));
    int w = __ffsll(b) - 1;
    avail &= ~(1u << w);
    if (lane == w) sel = 1;
    sum += v;
  }
  combine[(size_t)t * N_EXP + lane] = sel ? (s / (sum + 1e-20f)) : 0.f;
}

// =====================================================================
// Kernel 1: act[e][t][f] = combine[t][e] * silu(x@wg_e) * (x@wu_e)  (bf16)
// grid (F_EXP/128, T/128, N_EXP), block 256
// =====================================================================
__global__ __launch_bounds__(256) void moe_act_kernel(const float* __restrict__ x,
                                                      const float* __restrict__ wg,
                                                      const float* __restrict__ wu,
                                                      const float* __restrict__ combine,
                                                      BF16* __restrict__ act) {
  __shared__ BF16 As[128 * 32];
  __shared__ BF16 Bg[128 * 32];
  __shared__ BF16 Bu[128 * 32];

  const int fT = blockIdx.x, tT = blockIdx.y, e = blockIdx.z;
  const int wid = threadIdx.x >> 5;
  const int wm  = (wid >> 1) * 32;
  const int wn  = (wid & 1) * 64;

  const float* xa  = x  + (size_t)(tT * 128) * HID;
  const float* wge = wg + (size_t)e * HID * F_EXP + fT * 128;
  const float* wue = wu + (size_t)e * HID * F_EXP + fT * 128;

  v8f zf = {0.f, 0.f, 0.f, 0.f, 0.f, 0.f, 0.f, 0.f};
  v8f accg[2][4], accu[2][4];
#pragma unroll
  for (int mi = 0; mi < 2; ++mi)
#pragma unroll
    for (int n = 0; n < 4; ++n) { accg[mi][n] = zf; accu[mi][n] = zf; }

  for (int k0 = 0; k0 < HID; k0 += 32) {
    __syncthreads();
    g2l_f32  (As, xa + k0, HID);
    g2l_f32_T(Bg, wge + (size_t)k0 * F_EXP, F_EXP);
    g2l_f32_T(Bu, wue + (size_t)k0 * F_EXP, F_EXP);
    if (k0 + 32 < HID) {
      __builtin_prefetch(wge + (size_t)(k0 + 32) * F_EXP, 0, 1);
      __builtin_prefetch(wue + (size_t)(k0 + 32) * F_EXP, 0, 1);
    }
    __syncthreads();

    v16bf a0 = frag_ld(As, wm);
    v16bf a1 = frag_ld(As, wm + 16);
#pragma unroll
    for (int n = 0; n < 4; ++n) {
      v16bf bg = frag_ld(Bg, wn + n * 16);
      accg[0][n] = wmma_bf16(a0, bg, accg[0][n]);
      accg[1][n] = wmma_bf16(a1, bg, accg[1][n]);
      v16bf bu = frag_ld(Bu, wn + n * 16);
      accu[0][n] = wmma_bf16(a0, bu, accu[0][n]);
      accu[1][n] = wmma_bf16(a1, bu, accu[1][n]);
    }
  }

  const int rowBase = tT * 128 + wm;
  const int colBase = fT * 128 + wn;
#pragma unroll
  for (int mi = 0; mi < 2; ++mi)
#pragma unroll
    for (int n = 0; n < 4; ++n)
#pragma unroll
      for (int i = 0; i < 8; ++i) {
        int m = rowBase + mi * 16 + c_row(i);
        int c = colBase + n * 16 + c_col();
        float w = combine[(size_t)m * N_EXP + e];
        float a = silu(accg[mi][n][i]) * accu[mi][n][i] * w;
        act[((size_t)e * T_TOK + m) * F_EXP + c] = to_bf16(a);
      }
}

// =====================================================================
// Kernel 2: routed += sum_e act_e @ wd_e  (split-K over experts, atomics)
// grid (HID/128, T/128, 4), block 256; A-panel staged via TDM
// =====================================================================
__global__ __launch_bounds__(256) void moe_down_kernel(const BF16* __restrict__ act,
                                                       const float* __restrict__ wd,
                                                       float* __restrict__ out) {
  __shared__ BF16 As[128 * 32];
  __shared__ BF16 Bs[128 * 32];

  const int dT = blockIdx.x, tT = blockIdx.y, ez = blockIdx.z;
  const int wid = threadIdx.x >> 5;
  const int wm  = (wid >> 1) * 32;
  const int wn  = (wid & 1) * 64;

  v8f zf = {0.f, 0.f, 0.f, 0.f, 0.f, 0.f, 0.f, 0.f};
  v8f acc[2][4];
#pragma unroll
  for (int mi = 0; mi < 2; ++mi)
#pragma unroll
    for (int n = 0; n < 4; ++n) acc[mi][n] = zf;

  for (int e = ez * 8; e < ez * 8 + 8; ++e) {
    const BF16*  ae = act + ((size_t)e * T_TOK + tT * 128) * F_EXP;
    const float* we = wd  + (size_t)e * F_EXP * HID + dT * 128;
    for (int k0 = 0; k0 < F_EXP; k0 += 32) {
      __syncthreads();
      if (threadIdx.x == 0)
        tdm_load_bf16_tile(As, ae + k0, F_EXP, 128, 128);   // [128 rows x 32 k] panel
      g2l_f32_T(Bs, we + (size_t)k0 * HID, HID);
      if (k0 + 32 < F_EXP) __builtin_prefetch(we + (size_t)(k0 + 32) * HID, 0, 1);
      if (wid == 0) __builtin_amdgcn_s_wait_tensorcnt(0);
      __syncthreads();

      v16bf a0 = frag_ld(As, wm);
      v16bf a1 = frag_ld(As, wm + 16);
#pragma unroll
      for (int n = 0; n < 4; ++n) {
        v16bf b = frag_ld(Bs, wn + n * 16);
        acc[0][n] = wmma_bf16(a0, b, acc[0][n]);
        acc[1][n] = wmma_bf16(a1, b, acc[1][n]);
      }
    }
  }

  const int rowBase = tT * 128 + wm;
  const int colBase = dT * 128 + wn;
#pragma unroll
  for (int mi = 0; mi < 2; ++mi)
#pragma unroll
    for (int n = 0; n < 4; ++n)
#pragma unroll
      for (int i = 0; i < 8; ++i) {
        int m = rowBase + mi * 16 + c_row(i);
        int c = colBase + n * 16 + c_col();
        atomicAdd(&out[(size_t)m * HID + c], acc[mi][n][i]);
      }
}

// =====================================================================
// Kernel 3: sact = silu(x@sg) * (x@su)  (bf16)
// grid (F_SH/128, T/128), block 256
// =====================================================================
__global__ __launch_bounds__(256) void shared_act_kernel(const float* __restrict__ x,
                                                         const float* __restrict__ sg,
                                                         const float* __restrict__ su,
                                                         BF16* __restrict__ sact) {
  __shared__ BF16 As[128 * 32];
  __shared__ BF16 Bg[128 * 32];
  __shared__ BF16 Bu[128 * 32];

  const int fT = blockIdx.x, tT = blockIdx.y;
  const int wid = threadIdx.x >> 5;
  const int wm  = (wid >> 1) * 32;
  const int wn  = (wid & 1) * 64;

  const float* xa = x + (size_t)(tT * 128) * HID;

  v8f zf = {0.f, 0.f, 0.f, 0.f, 0.f, 0.f, 0.f, 0.f};
  v8f accg[2][4], accu[2][4];
#pragma unroll
  for (int mi = 0; mi < 2; ++mi)
#pragma unroll
    for (int n = 0; n < 4; ++n) { accg[mi][n] = zf; accu[mi][n] = zf; }

  for (int k0 = 0; k0 < HID; k0 += 32) {
    __syncthreads();
    g2l_f32  (As, xa + k0, HID);
    g2l_f32_T(Bg, sg + (size_t)k0 * F_SH + fT * 128, F_SH);
    g2l_f32_T(Bu, su + (size_t)k0 * F_SH + fT * 128, F_SH);
    __syncthreads();

    v16bf a0 = frag_ld(As, wm);
    v16bf a1 = frag_ld(As, wm + 16);
#pragma unroll
    for (int n = 0; n < 4; ++n) {
      v16bf bg = frag_ld(Bg, wn + n * 16);
      accg[0][n] = wmma_bf16(a0, bg, accg[0][n]);
      accg[1][n] = wmma_bf16(a1, bg, accg[1][n]);
      v16bf bu = frag_ld(Bu, wn + n * 16);
      accu[0][n] = wmma_bf16(a0, bu, accu[0][n]);
      accu[1][n] = wmma_bf16(a1, bu, accu[1][n]);
    }
  }

  const int rowBase = tT * 128 + wm;
  const int colBase = fT * 128 + wn;
#pragma unroll
  for (int mi = 0; mi < 2; ++mi)
#pragma unroll
    for (int n = 0; n < 4; ++n)
#pragma unroll
      for (int i = 0; i < 8; ++i) {
        int m = rowBase + mi * 16 + c_row(i);
        int c = colBase + n * 16 + c_col();
        sact[(size_t)m * F_SH + c] = to_bf16(silu(accg[mi][n][i]) * accu[mi][n][i]);
      }
}

// =====================================================================
// Kernel 4: out += sact @ sd   (atomic accumulate)
// grid (HID/128, T/128), block 256
// =====================================================================
__global__ __launch_bounds__(256) void shared_down_kernel(const BF16* __restrict__ sact,
                                                          const float* __restrict__ sd,
                                                          float* __restrict__ out) {
  __shared__ BF16 As[128 * 32];
  __shared__ BF16 Bs[128 * 32];

  const int dT = blockIdx.x, tT = blockIdx.y;
  const int wid = threadIdx.x >> 5;
  const int wm  = (wid >> 1) * 32;
  const int wn  = (wid & 1) * 64;

  const BF16* ae = sact + (size_t)(tT * 128) * F_SH;

  v8f zf = {0.f, 0.f, 0.f, 0.f, 0.f, 0.f, 0.f, 0.f};
  v8f acc[2][4];
#pragma unroll
  for (int mi = 0; mi < 2; ++mi)
#pragma unroll
    for (int n = 0; n < 4; ++n) acc[mi][n] = zf;

  for (int k0 = 0; k0 < F_SH; k0 += 32) {
    __syncthreads();
    g2l_bf16 (As, ae + k0, F_SH);
    g2l_f32_T(Bs, sd + (size_t)k0 * HID + dT * 128, HID);
    if (k0 + 32 < F_SH) __builtin_prefetch(sd + (size_t)(k0 + 32) * HID + dT * 128, 0, 1);
    __syncthreads();

    v16bf a0 = frag_ld(As, wm);
    v16bf a1 = frag_ld(As, wm + 16);
#pragma unroll
    for (int n = 0; n < 4; ++n) {
      v16bf b = frag_ld(Bs, wn + n * 16);
      acc[0][n] = wmma_bf16(a0, b, acc[0][n]);
      acc[1][n] = wmma_bf16(a1, b, acc[1][n]);
    }
  }

  const int rowBase = tT * 128 + wm;
  const int colBase = dT * 128 + wn;
#pragma unroll
  for (int mi = 0; mi < 2; ++mi)
#pragma unroll
    for (int n = 0; n < 4; ++n)
#pragma unroll
      for (int i = 0; i < 8; ++i) {
        int m = rowBase + mi * 16 + c_row(i);
        int c = colBase + n * 16 + c_col();
        atomicAdd(&out[(size_t)m * HID + c], acc[mi][n][i]);
      }
}

// =====================================================================
extern "C" void kernel_launch(void* const* d_in, const int* in_sizes, int n_in,
                              void* d_out, int out_size, void* d_ws, size_t ws_size,
                              hipStream_t stream) {
  const float* x   = (const float*)d_in[0];  // [1,1024,2048]
  const float* gw  = (const float*)d_in[1];  // [2048,32]
  const float* wg  = (const float*)d_in[2];  // [32,2048,768]
  const float* wu  = (const float*)d_in[3];  // [32,2048,768]
  const float* wd  = (const float*)d_in[4];  // [32,768,2048]
  const float* sg  = (const float*)d_in[5];  // [2048,1536]
  const float* su  = (const float*)d_in[6];  // [2048,1536]
  const float* sd  = (const float*)d_in[7];  // [1536,2048]
  float* out = (float*)d_out;                // [1,1024,2048]

  // workspace layout
  float* combine = (float*)d_ws;                                     // 1024*32 f32
  BF16*  act     = (BF16*)((char*)d_ws + (size_t)T_TOK * N_EXP * 4); // 32*1024*768 bf16
  BF16*  sact    = act + (size_t)N_EXP * T_TOK * F_EXP;              // 1024*1536 bf16

  // zero the output (both down-projections accumulate with float atomics)
  hipMemsetAsync(out, 0, (size_t)out_size * sizeof(float), stream);

  // 0) gating
  gate_kernel<<<dim3(T_TOK / 4), dim3(128), 0, stream>>>(x, gw, combine);

  // 1) per-expert up/gate + SwiGLU + combine scale (dense over all experts)
  moe_act_kernel<<<dim3(F_EXP / 128, T_TOK / 128, N_EXP), dim3(256), 0, stream>>>(
      x, wg, wu, combine, act);

  // 2) routed += sum_e act_e @ wd_e  (split-K over experts)
  moe_down_kernel<<<dim3(HID / 128, T_TOK / 128, 4), dim3(256), 0, stream>>>(act, wd, out);

  // 3) shared-expert SwiGLU
  shared_act_kernel<<<dim3(F_SH / 128, T_TOK / 128), dim3(256), 0, stream>>>(x, sg, su, sact);

  // 4) out += sact @ sd
  shared_down_kernel<<<dim3(HID / 128, T_TOK / 128), dim3(256), 0, stream>>>(sact, sd, out);
}
